// SelfAttention_83975200572056
// MI455X (gfx1250) — matrix operand order repaired and checked by reading, not compiled, not run on hardware
//
#include <hip/hip_runtime.h>

typedef __bf16 bf16;
typedef __attribute__((ext_vector_type(16))) __bf16 v16bf;
typedef __attribute__((ext_vector_type(8)))  __bf16 v8bf;
typedef __attribute__((ext_vector_type(8)))  float  v8f;

#define B_  2
#define S_  2048
#define C_  512
#define E_  512
#define H_  8
#define HD_ 64
#define M_  (B_ * S_)            // 4096
#define SCALE_ 0.125f            // HD^-0.5

// ---------------------------------------------------------------------------
// WMMA helpers (CDNA5 / gfx1250, wave32).
// A fragment: 16x32 bf16.  k order used by this code (contiguous, not the ISA table order):
//   lane<16 : row M=lane,    K = 0..15 contiguous (elem e -> K=e)
//   lane>=16: row M=lane-16, K = 16..31 contiguous
// B fragment: 32x16 bf16 (K x N), B[k][n] = base[n*ld + k]:
//   lane<16 : col N=lane,    K = 0..15 contiguous (elem e -> K=e)
//   lane>=16: col N=lane-16, K = 16..31 contiguous
// C/D 16x16 f32: acc[r] = element (M = r + 8*(lane>=16), N = lane&15)
// ---------------------------------------------------------------------------

__device__ __forceinline__ v16bf load_a_frag(const bf16* base, int ld, int lane) {
  int m  = lane & 15;
  int k0 = (lane < 16) ? 0 : 16;
  const bf16* p = base + (size_t)m * ld + k0;
  v8bf lo = *(const v8bf*)(p);        // K = k0 .. k0+7
  v8bf hi = *(const v8bf*)(p + 8);    // K = k0+8 .. k0+15
  v16bf r;
#pragma unroll
  for (int i = 0; i < 8; ++i) { r[i] = lo[i]; r[i + 8] = hi[i]; }
  return r;
}

__device__ __forceinline__ v16bf load_b_frag(const bf16* base, int ld, int lane) {
  int n  = lane & 15;
  int kb = (lane < 16) ? 0 : 16;
  const bf16* p = base + (size_t)n * ld + kb;
  v8bf lo = *(const v8bf*)(p);        // K = kb .. kb+7
  v8bf hi = *(const v8bf*)(p + 8);    // K = kb+8 .. kb+15
  v16bf r;
#pragma unroll
  for (int i = 0; i < 8; ++i) { r[i] = lo[i]; r[i + 8] = hi[i]; }
  return r;
}

__device__ __forceinline__ v8f wmma_bf16(v16bf a, v16bf b, v8f c) {
  return __builtin_amdgcn_wmma_f32_16x16x32_bf16(false, a, false, b, (short)0, c,
                                                 false, false);
}

// ---------------------------------------------------------------------------
// Elementwise fp32 -> bf16 cast
// ---------------------------------------------------------------------------
__global__ void cast_f32_bf16_kernel(const float* __restrict__ src,
                                     bf16* __restrict__ dst, int n) {
  int i = blockIdx.x * blockDim.x + threadIdx.x;
  if (i < n) dst[i] = (bf16)src[i];
}

// ---------------------------------------------------------------------------
// V transpose: Vb viewed [B,H,S,HD] -> Vt [B,H,HD,S]  (write-coalesced)
// ---------------------------------------------------------------------------
__global__ void transpose_v_kernel(const bf16* __restrict__ Vb,
                                   bf16* __restrict__ Vt) {
  size_t i = (size_t)blockIdx.x * blockDim.x + threadIdx.x;
  size_t total = (size_t)B_ * H_ * HD_ * S_;
  if (i >= total) return;
  int    j  = (int)(i & (S_ - 1));
  size_t r  = i >> 11;                 // S_ = 2^11
  int    d  = (int)(r & (HD_ - 1));
  size_t bh = r >> 6;                  // HD_ = 2^6
  Vt[i] = Vb[bh * (size_t)(S_ * HD_) + (size_t)j * HD_ + d];
}

// ---------------------------------------------------------------------------
// NT GEMM: Y[4096,512] = A[4096,512](bf16) * W[512,512]^T(bf16) + bias.
// K = 512 is compile-time: fully-unrolled, register double-buffered k-loop
// (buffers become SSA -> no rotate moves, staggered s_wait_loadcnt).
// Block = 256 thr (8 waves): 4 m-tiles x 2 n-strips; wave = 16m x 64n.
// Grid = (4096/64, 512/128) = (64, 4).
// ---------------------------------------------------------------------------
template <bool BF16OUT>
__global__ void gemm_nt_kernel(const bf16* __restrict__ A,
                               const bf16* __restrict__ W,
                               const float* __restrict__ bias,
                               bf16* __restrict__ outb,
                               float* __restrict__ outf) {
  constexpr int K = 512;
  constexpr int N = 512;
  int tid = threadIdx.x, lane = tid & 31, wave = tid >> 5;
  int mt = wave & 3, ns = wave >> 2;
  int m0 = blockIdx.x * 64 + mt * 16;
  int n0 = blockIdx.y * 128 + ns * 64;

  v8f acc[4] = {};
  v16bf a = load_a_frag(A + (size_t)m0 * K, K, lane);
  v16bf b[4];
#pragma unroll
  for (int t = 0; t < 4; ++t)
    b[t] = load_b_frag(W + (size_t)(n0 + t * 16) * K, K, lane);

#pragma unroll
  for (int k0 = 0; k0 < K; k0 += 32) {
    v16bf na = a;
    v16bf nb[4] = {b[0], b[1], b[2], b[3]};
    if (k0 + 32 < K) {
      na = load_a_frag(A + (size_t)m0 * K + k0 + 32, K, lane);
#pragma unroll
      for (int t = 0; t < 4; ++t)
        nb[t] = load_b_frag(W + (size_t)(n0 + t * 16) * K + k0 + 32, K, lane);
    }
#pragma unroll
    for (int t = 0; t < 4; ++t) acc[t] = wmma_bf16(a, b[t], acc[t]);
    a = na;
#pragma unroll
    for (int t = 0; t < 4; ++t) b[t] = nb[t];
  }

  int mb = (lane >> 4) << 3;
  int nc = lane & 15;
#pragma unroll
  for (int t = 0; t < 4; ++t) {
    int col = n0 + t * 16 + nc;
    float bv = bias[col];
#pragma unroll
    for (int r = 0; r < 8; ++r) {
      int row = m0 + r + mb;
      float v = acc[t][r] + bv;
      if (BF16OUT) outb[(size_t)row * N + col] = (bf16)v;
      else         outf[(size_t)row * N + col] = v;
    }
  }
}

// ---------------------------------------------------------------------------
// Fused attention: one block = 16 query rows of one (b,h). 8 waves, each wave
// owns a contiguous 256-wide key range.
// LDS: fp32 logits (128KB) + bf16 P (64KB) + reduction scratch + dist-staging
// double buffer (2 x 16KB, filled with GLOBAL_LOAD_ASYNC_TO_LDS).
// ctx partial-sum buffer aliases dead logits.
// ---------------------------------------------------------------------------
#define SMEM_LOGITS_BYTES (16 * 2048 * 4)                       // 131072
#define SMEM_PBF_BYTES    (16 * 2048 * 2)                       //  65536
#define SMEM_RED_BYTES    ((16 * 16 + 32) * 4)                  //   1152
#define SMEM_DIST_OFF     (SMEM_LOGITS_BYTES + SMEM_PBF_BYTES + SMEM_RED_BYTES)
#define SMEM_DIST_CHUNK   (16 * 256 * 4)                        //  16384
#define ATTN_SMEM_BYTES   (SMEM_DIST_OFF + 2 * SMEM_DIST_CHUNK) // 230528

// async-copy one 16x256 fp32 chunk of dist into LDS staging buffer `buf`
__device__ __forceinline__ void issue_dist_chunk(const float* dbh, int c,
                                                 unsigned dist_lds_off, int buf,
                                                 int tid) {
#pragma unroll
  for (int q = 0; q < 4; ++q) {
    int lin = (tid * 4 + q) * 4;           // float index within chunk
    int row = lin >> 8;                    // 256 floats per row
    int col = lin & 255;
    unsigned long long g =
        (unsigned long long)(dbh + (size_t)row * S_ + c * 256 + col);
    unsigned l = dist_lds_off + (unsigned)(buf * SMEM_DIST_CHUNK + lin * 4);
    asm volatile("global_load_async_to_lds_b128 %0, %1, off"
                 :: "v"(l), "v"(g) : "memory");
  }
}

__global__ void attn_kernel(const bf16* __restrict__ Qb,
                            const bf16* __restrict__ Kb,
                            const bf16* __restrict__ Vt,
                            const int*  __restrict__ mask,
                            const float* __restrict__ dist,
                            float* __restrict__ attn_out,
                            bf16* __restrict__ ctxb) {
  extern __shared__ char smem[];
  float* logits = (float*)smem;                                   // [16][2048]
  bf16*  pbf    = (bf16*)(smem + SMEM_LOGITS_BYTES);              // [16][2048]
  float* red    = (float*)(smem + SMEM_LOGITS_BYTES + SMEM_PBF_BYTES); // [16][16]
  float* rowmax = red + 16 * 16;
  float* rowsum = rowmax + 16;
  float* distbuf = (float*)(smem + SMEM_DIST_OFF);  // [2][16][256]
  float* ctxred  = logits;  // [8][16][64] — logits are dead by phase 4
  // LDS byte offset of the staging buffer (shared aperture = upper 32 bits)
  unsigned dist_lds_off = (unsigned)(unsigned long long)(smem + SMEM_DIST_OFF);

  int tid = threadIdx.x, lane = tid & 31, wave = tid >> 5;
  int bh = blockIdx.y;
  int b  = bh >> 3;
  int h  = bh & 7;
  int m0 = blockIdx.x * 16;

  const bf16*  Qbh = Qb + (size_t)bh * S_ * HD_;
  const bf16*  Kbh = Kb + (size_t)bh * S_ * HD_;
  const bf16*  Vth = Vt + (size_t)bh * HD_ * S_;
  const float* dbh = dist + (size_t)bh * S_ * S_ + (size_t)m0 * S_;
  float*       abh = attn_out + (size_t)bh * S_ * S_ + (size_t)m0 * S_;

  int mb = (lane >> 4) << 3;
  int nc = lane & 15;

  // ---- kick off dist_map prefetch (independent of all compute) ----
  issue_dist_chunk(dbh, 0, dist_lds_off, 0, tid);
  issue_dist_chunk(dbh, 1, dist_lds_off, 1, tid);

  // ---- phase 1: Q fragments + per-lane mask preload (one clause, no
  //      per-tile loadcnt-0 stalls in the WMMA loop) ----
  v16bf a0 = load_a_frag(Qbh + (size_t)m0 * HD_ + 0,  HD_, lane);
  v16bf a1 = load_a_frag(Qbh + (size_t)m0 * HD_ + 32, HD_, lane);
  int padv[16];
#pragma unroll
  for (int t = 0; t < 16; ++t)
    padv[t] = mask[b * S_ + wave * 256 + t * 16 + nc];

  // ---- phase 2: logits = mask(Q K^T) * SCALE into LDS ----
  // fully unrolled + double-buffered: loads of tile t+1 overlap WMMAs of t
  {
    int jbase = wave * 256;
    v16bf b0 = load_b_frag(Kbh + (size_t)jbase * HD_ + 0,  HD_, lane);
    v16bf b1 = load_b_frag(Kbh + (size_t)jbase * HD_ + 32, HD_, lane);
#pragma unroll
    for (int t = 0; t < 16; ++t) {
      int j0 = jbase + t * 16;
      v16bf nb0 = b0, nb1 = b1;
      if (t < 15) {
        nb0 = load_b_frag(Kbh + (size_t)(j0 + 16) * HD_ + 0,  HD_, lane);
        nb1 = load_b_frag(Kbh + (size_t)(j0 + 16) * HD_ + 32, HD_, lane);
      }
      v8f acc = {};
      acc = wmma_bf16(a0, b0, acc);
      acc = wmma_bf16(a1, b1, acc);
      int j = j0 + nc;
      bool pad = (padv[t] != 0);
#pragma unroll
      for (int r = 0; r < 8; ++r) {
        float v = pad ? -10000.0f : acc[r];
        logits[(r + mb) * 2048 + j] = v * SCALE_;
      }
      b0 = nb0; b1 = nb1;
    }
  }
  __syncthreads();

  // ---- phase 3a: row max / row sum (16 rows x 16 threads each) ----
  int row = tid >> 4, c16 = tid & 15;
  float mx = -3.4e38f;
  for (int i = 0; i < 128; ++i)
    mx = fmaxf(mx, logits[row * 2048 + c16 + i * 16]);
  red[row * 16 + c16] = mx;
  __syncthreads();
  if (c16 == 0) {
    float m2 = red[row * 16];
    for (int i = 1; i < 16; ++i) m2 = fmaxf(m2, red[row * 16 + i]);
    rowmax[row] = m2;
  }
  __syncthreads();
  float rm = rowmax[row];
  float sm = 0.f;
  for (int i = 0; i < 128; ++i)
    sm += __expf(logits[row * 2048 + c16 + i * 16] - rm);
  red[row * 16 + c16] = sm;
  __syncthreads();
  if (c16 == 0) {
    float s2 = 0.f;
    for (int i = 0; i < 16; ++i) s2 += red[row * 16 + i];
    rowsum[row] = s2;
  }
  __syncthreads();
  float inv = 1.0f / rowsum[row];

  // ---- phase 3b: softmax + dist blend, chunked over async-staged dist ----
  for (int c = 0; c < 8; ++c) {
    if (c == 7) asm volatile("s_wait_asynccnt 0x0" ::: "memory");
    else        asm volatile("s_wait_asynccnt 0x4" ::: "memory");
    __syncthreads();                        // chunk c visible to all waves
    const float* dch = distbuf + (c & 1) * (16 * 256);
#pragma unroll 4
    for (int ii = 0; ii < 16; ++ii) {
      int jl = c16 + ii * 16;               // column within chunk
      int j  = c * 256 + jl;
      float p  = __expf(logits[row * 2048 + j] - rm) * inv;
      float pf = (p + dch[row * 256 + jl]) * 0.5f;     // attn_probs
      abh[(size_t)row * S_ + j] = pf;
      pbf[row * 2048 + j] = (bf16)pf;
    }
    __syncthreads();                        // all done reading buffer (c&1)
    if (c + 2 < 8) issue_dist_chunk(dbh, c + 2, dist_lds_off, c & 1, tid);
  }
  __syncthreads();

  // ---- phase 4: ctx = P @ V (unrolled + double-buffered) ----
  v8f c4[4] = {};
  {
    int jbase = wave * 256;
    v16bf ap = load_a_frag(pbf + jbase, 2048, lane);
    v16bf vb[4];
#pragma unroll
    for (int dt = 0; dt < 4; ++dt)
      vb[dt] = load_b_frag(Vth + (size_t)(dt * 16) * S_ + jbase, S_, lane);
#pragma unroll
    for (int s8 = 0; s8 < 8; ++s8) {
      int j0 = jbase + s8 * 32;
      v16bf nap = ap;
      v16bf nvb[4] = {vb[0], vb[1], vb[2], vb[3]};
      if (s8 < 7) {
        nap = load_a_frag(pbf + j0 + 32, 2048, lane);
#pragma unroll
        for (int dt = 0; dt < 4; ++dt)
          nvb[dt] = load_b_frag(Vth + (size_t)(dt * 16) * S_ + j0 + 32, S_, lane);
      }
#pragma unroll
      for (int dt = 0; dt < 4; ++dt) c4[dt] = wmma_bf16(ap, vb[dt], c4[dt]);
      ap = nap;
#pragma unroll
      for (int dt = 0; dt < 4; ++dt) vb[dt] = nvb[dt];
    }
  }
#pragma unroll
  for (int dt = 0; dt < 4; ++dt)
#pragma unroll
    for (int r = 0; r < 8; ++r)
      ctxred[(wave * 16 + r + mb) * 64 + dt * 16 + nc] = c4[dt][r];
  __syncthreads();

  // cross-wave reduction; store ctx transposed into [B,S,E] bf16 layout
  for (int e = tid * 4; e < tid * 4 + 4; ++e) {
    int m = e >> 6, d = e & 63;
    float s = 0.f;
    for (int w = 0; w < 8; ++w) s += ctxred[(w * 16 + m) * 64 + d];
    ctxb[((size_t)(b * S_ + m0 + m)) * E_ + h * 64 + d] = (bf16)s;
  }
}

// ---------------------------------------------------------------------------
extern "C" void kernel_launch(void* const* d_in, const int* in_sizes, int n_in,
                              void* d_out, int out_size, void* d_ws, size_t ws_size,
                              hipStream_t stream) {
  const float* x    = (const float*)d_in[0];
  const float* dist = (const float*)d_in[1];
  const int*   mask = (const int*)d_in[2];
  const float* Wq   = (const float*)d_in[3];
  const float* bq   = (const float*)d_in[4];
  const float* Wk   = (const float*)d_in[5];
  const float* bk   = (const float*)d_in[6];
  const float* Wv   = (const float*)d_in[7];
  const float* bv   = (const float*)d_in[8];
  const float* Wo   = (const float*)d_in[9];
  const float* bo   = (const float*)d_in[10];

  float* out_f    = (float*)d_out;                       // [B,S,E]
  float* attn_out = out_f + (size_t)B_ * S_ * E_;        // [B,H,S,S]

  char* ws = (char*)d_ws;
  size_t off = 0;
  auto alloc = [&](size_t bytes) {
    char* p = ws + off;
    off += (bytes + 255) & ~(size_t)255;
    return p;
  };
  bf16* xb   = (bf16*)alloc((size_t)M_ * C_ * 2);
  bf16* Wqb  = (bf16*)alloc((size_t)E_ * C_ * 2);
  bf16* Wkb  = (bf16*)alloc((size_t)E_ * C_ * 2);
  bf16* Wvb  = (bf16*)alloc((size_t)E_ * C_ * 2);
  bf16* Wob  = (bf16*)alloc((size_t)E_ * E_ * 2);
  bf16* Qb   = (bf16*)alloc((size_t)M_ * E_ * 2);
  bf16* Kb   = (bf16*)alloc((size_t)M_ * E_ * 2);
  bf16* Vb   = (bf16*)alloc((size_t)M_ * E_ * 2);
  bf16* Vt   = (bf16*)alloc((size_t)M_ * E_ * 2);
  bf16* ctxb = (bf16*)alloc((size_t)M_ * E_ * 2);

  int n;
  n = M_ * C_;
  cast_f32_bf16_kernel<<<(n + 255) / 256, 256, 0, stream>>>(x, xb, n);
  n = E_ * C_;
  cast_f32_bf16_kernel<<<(n + 255) / 256, 256, 0, stream>>>(Wq, Wqb, n);
  cast_f32_bf16_kernel<<<(n + 255) / 256, 256, 0, stream>>>(Wk, Wkb, n);
  cast_f32_bf16_kernel<<<(n + 255) / 256, 256, 0, stream>>>(Wv, Wvb, n);
  cast_f32_bf16_kernel<<<(n + 255) / 256, 256, 0, stream>>>(Wo, Wob, n);

  dim3 gg(M_ / 64, E_ / 128);
  gemm_nt_kernel<true><<<gg, 256, 0, stream>>>(xb, Wqb, bq, Qb, nullptr);
  gemm_nt_kernel<true><<<gg, 256, 0, stream>>>(xb, Wkb, bk, Kb, nullptr);
  gemm_nt_kernel<true><<<gg, 256, 0, stream>>>(xb, Wvb, bv, Vb, nullptr);

  n = B_ * H_ * HD_ * S_;
  transpose_v_kernel<<<(n + 255) / 256, 256, 0, stream>>>(Vb, Vt);

  (void)hipFuncSetAttribute((const void*)attn_kernel,
                            hipFuncAttributeMaxDynamicSharedMemorySize,
                            ATTN_SMEM_BYTES);
  attn_kernel<<<dim3(S_ / 16, B_ * H_), 256, ATTN_SMEM_BYTES, stream>>>(
      Qb, Kb, Vt, mask, dist, attn_out, ctxb);

  gemm_nt_kernel<false><<<gg, 256, 0, stream>>>(ctxb, Wob, bo, nullptr, out_f);
}
